// SlotGrouping_65206193487897
// MI455X (gfx1250) — matrix operand order repaired and checked by reading, not compiled, not run on hardware
//
#include <hip/hip_runtime.h>
#include <hip/hip_bf16.h>
#include <math.h>

// ---------------- problem constants ----------------
#define BB   32
#define DD   256
#define HWW  4096
#define KK   256
#define DFFN 2048
#define BKK  (BB * KK)

#define TEMP_INV (1.0f / 0.07f)
#define EPS_ATTN 1e-6f
#define NORM_EPS 1e-12f
#define LN_EPS   1e-5f

typedef __bf16 bf16_t;
typedef __bf16 v16bf __attribute__((ext_vector_type(16)));
typedef __bf16 v8bf  __attribute__((ext_vector_type(8)));
typedef float  v8f   __attribute__((ext_vector_type(8)));

__device__ __forceinline__ v8f zero8() {
  v8f z = {0.f, 0.f, 0.f, 0.f, 0.f, 0.f, 0.f, 0.f};
  return z;
}

// D = A(16x32 bf16) * B(32x16 bf16) + C(16x16 f32)
__device__ __forceinline__ v8f wmma_bf16(v16bf a, v16bf b, v8f c) {
  return __builtin_amdgcn_wmma_f32_16x16x32_bf16(
      /*neg_a=*/false, a, /*neg_b=*/false, b,
      /*c_mod=*/(short)0, c, /*reuse_a=*/false, /*reuse_b=*/false);
}

// Async 16B global -> LDS copy (per-lane). ASYNCcnt tracked.
// LDS offset = low 32 bits of the flat shared address (flat shared = {aperture, lds_off}).
__device__ __forceinline__ void async_load_16B(void* lds_dst, const void* gsrc) {
  unsigned int loff = (unsigned int)(uintptr_t)lds_dst;
  asm volatile("global_load_async_to_lds_b128 %0, %1, off"
               :
               : "v"(loff), "v"(gsrc)
               : "memory");
}
__device__ __forceinline__ void wait_async_zero() {
  asm volatile("s_wait_asynccnt 0" ::: "memory");
}

// A fragment: 16x32 tile, row-major, leading dim lda (elements).
// lane<16: row=lane,    K = {0..15}
// lane>=16: row=lane-16, K = {16..31}
__device__ __forceinline__ v16bf load_a_frag(const bf16_t* base, int lda, int lane) {
  const int m  = lane & 15;
  const int k0 = (lane >> 4) << 4;  // 0 or 16
  const bf16_t* p = base + (size_t)m * lda + k0;
  v8bf lo = *(const v8bf*)(p);
  v8bf hi = *(const v8bf*)(p + 8);
  v16bf a;
#pragma unroll
  for (int i = 0; i < 8; ++i) { a[i] = lo[i]; a[8 + i] = hi[i]; }
  return a;
}

// B fragment: 32x16 tile stored K-contiguous (row = n, col = k), leading dim ldb.
// lane holds column n=lane%16, 16 contiguous K starting at 16*(lane/16).
__device__ __forceinline__ v16bf load_b_frag(const bf16_t* base, int ldb, int lane) {
  const int n  = lane & 15;
  const int k0 = (lane >> 4) << 4;  // 0 or 16
  return *(const v16bf*)(base + (size_t)n * ldb + k0);
}

// Same as load_b_frag but source is f32 (converted on the fly).
__device__ __forceinline__ v16bf load_b_frag_f32(const float* base, int ldb, int lane) {
  const int n  = lane & 15;
  const int k0 = (lane >> 4) << 4;
  const float* p = base + (size_t)n * ldb + k0;
  v16bf r;
#pragma unroll
  for (int i = 0; i < 16; ++i) r[i] = (bf16_t)p[i];
  return r;
}

// ---------------- prep kernels ----------------

// w1 (DFF,D) and w2 (D,DFF) -> bf16 (both already K-contiguous for their GEMM)
__global__ void slotgrp_cvt_w(const float* __restrict__ w1, const float* __restrict__ w2,
                              bf16_t* __restrict__ w1b, bf16_t* __restrict__ w2b) {
  int i = blockIdx.x * blockDim.x + threadIdx.x;
  if (i < DFFN * DD) {
    w1b[i] = (bf16_t)w1[i];
    w2b[i] = (bf16_t)w2[i];
  }
}

// tn = l2norm(tgt, axis=D) -> bf16, one wave per row
__global__ void slotgrp_prep_tgt(const float* __restrict__ tgt, bf16_t* __restrict__ tnb) {
  const int wave = threadIdx.x >> 5, lane = threadIdx.x & 31;
  const int row = blockIdx.x * 8 + wave;           // < B*K
  const float* src = tgt + (size_t)row * DD;
  float v[8];
  float ss = 0.f;
#pragma unroll
  for (int i = 0; i < 8; ++i) { v[i] = src[lane + 32 * i]; ss += v[i] * v[i]; }
#pragma unroll
  for (int o = 16; o > 0; o >>= 1) ss += __shfl_xor(ss, o, 32);
  const float sc = 1.0f / fmaxf(sqrtf(ss), NORM_EPS);
#pragma unroll
  for (int i = 0; i < 8; ++i) tnb[(size_t)row * DD + lane + 32 * i] = (bf16_t)(v[i] * sc);
}

// xinv[b,hw] = 1 / max(||x[b,:,hw]||, eps)   (coalesced over hw)
__global__ void slotgrp_xinv(const float* __restrict__ x, float* __restrict__ xinv) {
  const int b = blockIdx.x;
  const int hw = blockIdx.y * 256 + threadIdx.x;
  const float* px = x + (size_t)b * DD * HWW + hw;
  float ss = 0.f;
  for (int d = 0; d < DD; ++d) { float v = px[(size_t)d * HWW]; ss += v * v; }
  xinv[(size_t)b * HWW + hw] = 1.0f / fmaxf(sqrtf(ss), NORM_EPS);
}

// xtb[b][hw][d] = bf16( x[b][d][hw] * xinv[b][hw] )  via LDS 32x32 transpose
__global__ void slotgrp_xtb(const float* __restrict__ x, const float* __restrict__ xinv,
                            bf16_t* __restrict__ xtb) {
  __shared__ float tile[32][33];
  const int b = blockIdx.x, hw0 = blockIdx.y * 32, d0 = blockIdx.z * 32;
  const int tx = threadIdx.x & 31, ty = threadIdx.x >> 5;  // 32 x 8
  for (int r = ty; r < 32; r += 8)
    tile[r][tx] = x[(size_t)b * DD * HWW + (size_t)(d0 + r) * HWW + hw0 + tx];
  __syncthreads();
  for (int r = ty; r < 32; r += 8) {
    const int hw = hw0 + r;
    const float s = xinv[(size_t)b * HWW + hw];
    xtb[((size_t)b * HWW + hw) * DD + d0 + tx] = (bf16_t)(tile[tx][r] * s);
  }
}

// ---------------- fused dots + softmax(K) + spatial-sum ----------------
// Workgroup = (b, 32-wide hw tile). 8 waves, wave w owns rows [32w,32w+32).
// The shared B tile (32 hw rows x 256 d, 16 KB) is staged into LDS once per
// workgroup via async global->LDS copies (ASYNCcnt), removing the 8x redundant
// per-wave global loads of the shared operand.
__global__ void slotgrp_dots_softmax(const bf16_t* __restrict__ tnb,
                                     const bf16_t* __restrict__ xtb,
                                     bf16_t* __restrict__ attnb,
                                     float* __restrict__ S) {
  __shared__ float sm[KK][33];
  __shared__ bf16_t sB[32][264];   // row pad: 528 B -> +4 banks per row, conflict-free
  __shared__ float cred[32][8];
  __shared__ float ctot[32];
  const int b = blockIdx.x;
  const int hw0 = blockIdx.y * 32;
  const int tid = threadIdx.x;
  const int wave = tid >> 5, lane = tid & 31;
  const int m0 = wave * 32;

  // ---- async stage of B tile: 32 rows x 512 B, rows contiguous in xtb ----
  {
    const char* gB = (const char*)(xtb + ((size_t)b * HWW + hw0) * DD);  // 16384 contiguous bytes
#pragma unroll
    for (int i = 0; i < 4; ++i) {
      const int idx = i * 4096 + tid * 16;  // byte index within tile
      const int row = idx >> 9;             // /512
      const int col = idx & 511;
      async_load_16B((char*)&sB[row][0] + col, gB + idx);
    }
    wait_async_zero();
  }
  __syncthreads();

  v8f acc[2][2];
#pragma unroll
  for (int i = 0; i < 2; ++i)
#pragma unroll
    for (int j = 0; j < 2; ++j) acc[i][j] = zero8();

  const bf16_t* Ab = tnb + (size_t)b * KK * DD;
#pragma unroll 2
  for (int d0 = 0; d0 < DD; d0 += 32) {
    v16bf a0 = load_a_frag(Ab + (size_t)m0 * DD + d0, DD, lane);
    v16bf a1 = load_a_frag(Ab + (size_t)(m0 + 16) * DD + d0, DD, lane);
    v16bf b0 = load_b_frag(&sB[0][d0], 264, lane);
    v16bf b1 = load_b_frag(&sB[16][d0], 264, lane);
    acc[0][0] = wmma_bf16(a0, b0, acc[0][0]);
    acc[0][1] = wmma_bf16(a0, b1, acc[0][1]);
    acc[1][0] = wmma_bf16(a1, b0, acc[1][0]);
    acc[1][1] = wmma_bf16(a1, b1, acc[1][1]);
  }
  const int cm = (lane >> 4) * 8, cn = lane & 15;
#pragma unroll
  for (int i = 0; i < 2; ++i)
#pragma unroll
    for (int j = 0; j < 2; ++j)
#pragma unroll
      for (int r = 0; r < 8; ++r)
        sm[m0 + i * 16 + cm + r][j * 16 + cn] = acc[i][j][r];
  __syncthreads();

  // softmax over K (256 rows) per column; 8 threads per column
  const int col = tid & 31, q = tid >> 5;
  float mx = -3.4e38f;
  for (int r = q * 32; r < q * 32 + 32; ++r) mx = fmaxf(mx, sm[r][col]);
  cred[col][q] = mx;
  __syncthreads();
  float mtot = cred[col][0];
#pragma unroll
  for (int i = 1; i < 8; ++i) mtot = fmaxf(mtot, cred[col][i]);
  mtot *= TEMP_INV;
  __syncthreads();
  float s = 0.f;
  for (int r = q * 32; r < q * 32 + 32; ++r) {
    float e = __expf(sm[r][col] * TEMP_INV - mtot);
    sm[r][col] = e;
    s += e;
  }
  cred[col][q] = s;
  __syncthreads();
  if (q == 0) {
    float tt = 0.f;
#pragma unroll
    for (int i = 0; i < 8; ++i) tt += cred[col][i];
    ctot[col] = tt;
  }
  __syncthreads();

  // row pass: attn = e/sum + EPS, bf16 store, spatial-sum atomics
  const int k = tid;  // 0..255
  float rs = 0.f;
  bf16_t* arow = attnb + ((size_t)b * KK + k) * HWW + hw0;
  for (int c = 0; c < 32; ++c) {
    float p = sm[k][c] / ctot[c] + EPS_ATTN;
    rs += p;
    arow[c] = (bf16_t)p;
  }
  atomicAdd(&S[b * KK + k], rs);
}

// ---------------- tgt2 = (attn @ x^T) / S  + tgt  ----------------
__global__ void slotgrp_tgt2(const bf16_t* __restrict__ attnb, const float* __restrict__ x,
                             const float* __restrict__ S, const float* __restrict__ tgt,
                             float* __restrict__ u) {
  const int b = blockIdx.x;
  const int k0 = blockIdx.y * 64;
  const int d0 = blockIdx.z * 64;
  const int tid = threadIdx.x, wave = tid >> 5, lane = tid & 31;
  const int mt = wave & 3;
  const int n0 = (wave >> 2) * 32;
  v8f acc0 = zero8(), acc1 = zero8();
  const bf16_t* Ab = attnb + ((size_t)b * KK + k0 + mt * 16) * HWW;
  const float* Xb = x + ((size_t)b * DD + d0 + n0) * HWW;
  for (int hw = 0; hw < HWW; hw += 32) {
    __builtin_prefetch(Ab + hw + 256, 0, 1);  // global_prefetch_b8
    v16bf a  = load_a_frag(Ab + hw, HWW, lane);
    v16bf b0 = load_b_frag_f32(Xb + hw, HWW, lane);
    v16bf b1 = load_b_frag_f32(Xb + (size_t)16 * HWW + hw, HWW, lane);
    acc0 = wmma_bf16(a, b0, acc0);
    acc1 = wmma_bf16(a, b1, acc1);
  }
  const int cm = (lane >> 4) * 8, cn = lane & 15;
#pragma unroll
  for (int r = 0; r < 8; ++r) {
    const int m = k0 + mt * 16 + cm + r;
    const float inv = 1.0f / S[b * KK + m];
    const int n = d0 + n0 + cn;
    const size_t o = ((size_t)b * KK + m) * DD;
    u[o + n]      = acc0[r] * inv + tgt[o + n];
    u[o + n + 16] = acc1[r] * inv + tgt[o + n + 16];
  }
}

// ---------------- layernorm 1 (row per block) ----------------
__global__ void slotgrp_ln1(const float* __restrict__ u, const float* __restrict__ g,
                            const float* __restrict__ be, float* __restrict__ t,
                            bf16_t* __restrict__ tb) {
  __shared__ float red[256];
  const int row = blockIdx.x, tid = threadIdx.x;
  const float v = u[(size_t)row * DD + tid];
  red[tid] = v;
  __syncthreads();
  for (int s = 128; s > 0; s >>= 1) { if (tid < s) red[tid] += red[tid + s]; __syncthreads(); }
  const float mu = red[0] * (1.0f / DD);
  __syncthreads();
  const float dv = v - mu;
  red[tid] = dv * dv;
  __syncthreads();
  for (int s = 128; s > 0; s >>= 1) { if (tid < s) red[tid] += red[tid + s]; __syncthreads(); }
  const float var = red[0] * (1.0f / DD);
  const float o = dv * rsqrtf(var + LN_EPS) * g[tid] + be[tid];
  t[(size_t)row * DD + tid] = o;
  tb[(size_t)row * DD + tid] = (bf16_t)o;
}

// ---------------- FFN1: h = relu(t @ w1^T + b1), bf16 ----------------
__global__ void slotgrp_ffn1(const bf16_t* __restrict__ tb, const bf16_t* __restrict__ w1b,
                             const float* __restrict__ b1, bf16_t* __restrict__ h) {
  const int r0 = blockIdx.x * 64, f0 = blockIdx.y * 64;
  const int tid = threadIdx.x, wave = tid >> 5, lane = tid & 31;
  const int mt = wave & 3;
  const int n0 = (wave >> 2) * 32;
  v8f acc0 = zero8(), acc1 = zero8();
  const bf16_t* Ab = tb + (size_t)(r0 + mt * 16) * DD;
  const bf16_t* Bb = w1b + (size_t)(f0 + n0) * DD;
#pragma unroll 2
  for (int d0 = 0; d0 < DD; d0 += 32) {
    v16bf a  = load_a_frag(Ab + d0, DD, lane);
    v16bf b0 = load_b_frag(Bb + d0, DD, lane);
    v16bf b1f = load_b_frag(Bb + (size_t)16 * DD + d0, DD, lane);
    acc0 = wmma_bf16(a, b0, acc0);
    acc1 = wmma_bf16(a, b1f, acc1);
  }
  const int cm = (lane >> 4) * 8, cn = lane & 15;
#pragma unroll
  for (int r = 0; r < 8; ++r) {
    const int m = r0 + mt * 16 + cm + r;
    const int n = f0 + n0 + cn;
    const float v0 = fmaxf(acc0[r] + b1[n], 0.0f);
    const float v1 = fmaxf(acc1[r] + b1[n + 16], 0.0f);
    h[(size_t)m * DFFN + n]      = (bf16_t)v0;
    h[(size_t)m * DFFN + n + 16] = (bf16_t)v1;
  }
}

// ---------------- FFN2 + residual + layernorm 2 ----------------
__global__ void slotgrp_ffn2_ln(const bf16_t* __restrict__ h, const bf16_t* __restrict__ w2b,
                                const float* __restrict__ b2, const float* __restrict__ t,
                                const float* __restrict__ g, const float* __restrict__ be,
                                float* __restrict__ out) {
  __shared__ float y[32][257];
  __shared__ float red[32][8];
  const int r0 = blockIdx.x * 32;
  const int tid = threadIdx.x, wave = tid >> 5, lane = tid & 31;
  const int n0 = wave * 32;
  v8f acc[2][2];
#pragma unroll
  for (int i = 0; i < 2; ++i)
#pragma unroll
    for (int j = 0; j < 2; ++j) acc[i][j] = zero8();
  for (int ff = 0; ff < DFFN; ff += 32) {
    v16bf a0 = load_a_frag(h + (size_t)r0 * DFFN + ff, DFFN, lane);
    v16bf a1 = load_a_frag(h + (size_t)(r0 + 16) * DFFN + ff, DFFN, lane);
    v16bf b0 = load_b_frag(w2b + (size_t)n0 * DFFN + ff, DFFN, lane);
    v16bf b1 = load_b_frag(w2b + (size_t)(n0 + 16) * DFFN + ff, DFFN, lane);
    acc[0][0] = wmma_bf16(a0, b0, acc[0][0]);
    acc[0][1] = wmma_bf16(a0, b1, acc[0][1]);
    acc[1][0] = wmma_bf16(a1, b0, acc[1][0]);
    acc[1][1] = wmma_bf16(a1, b1, acc[1][1]);
  }
  const int cm = (lane >> 4) * 8, cn = lane & 15;
#pragma unroll
  for (int i = 0; i < 2; ++i)
#pragma unroll
    for (int j = 0; j < 2; ++j)
#pragma unroll
      for (int r = 0; r < 8; ++r) {
        const int ml = i * 16 + cm + r;
        const int n = n0 + j * 16 + cn;
        y[ml][n] = acc[i][j][r] + b2[n] + t[(size_t)(r0 + ml) * DD + n];
      }
  __syncthreads();

  // LN over D=256, 8 threads per row
  const int rid = tid >> 3, sub = tid & 7;
  float s = 0.f;
  for (int c = sub * 32; c < sub * 32 + 32; ++c) s += y[rid][c];
  red[rid][sub] = s;
  __syncthreads();
  float mu = 0.f;
#pragma unroll
  for (int i = 0; i < 8; ++i) mu += red[rid][i];
  mu *= (1.0f / DD);
  __syncthreads();
  float vs = 0.f;
  for (int c = sub * 32; c < sub * 32 + 32; ++c) { float d = y[rid][c] - mu; vs += d * d; }
  red[rid][sub] = vs;
  __syncthreads();
  float var = 0.f;
#pragma unroll
  for (int i = 0; i < 8; ++i) var += red[rid][i];
  var *= (1.0f / DD);
  const float rstd = rsqrtf(var + LN_EPS);
  for (int c = sub * 32; c < sub * 32 + 32; ++c)
    out[(size_t)(r0 + rid) * DD + c] = (y[rid][c] - mu) * rstd * g[c] + be[c];
}

// ---------------- launcher ----------------
extern "C" void kernel_launch(void* const* d_in, const int* in_sizes, int n_in,
                              void* d_out, int out_size, void* d_ws, size_t ws_size,
                              hipStream_t stream) {
  const float* x   = (const float*)d_in[0];
  const float* tgt = (const float*)d_in[1];
  const float* w1  = (const float*)d_in[2];
  const float* b1  = (const float*)d_in[3];
  const float* w2  = (const float*)d_in[4];
  const float* b2  = (const float*)d_in[5];
  const float* g2  = (const float*)d_in[6];
  const float* be2 = (const float*)d_in[7];
  const float* g3  = (const float*)d_in[8];
  const float* be3 = (const float*)d_in[9];
  float* out = (float*)d_out;

  char* w = (char*)d_ws;
  size_t off = 0;
  auto take = [&](size_t bytes) { char* p = w + off; off += (bytes + 255) & ~(size_t)255; return p; };
  bf16_t* tnb   = (bf16_t*)take((size_t)BKK * DD * 2);        // 4 MiB
  float*  xinv  = (float*)take((size_t)BB * HWW * 4);         // 512 KiB
  float*  S     = (float*)take((size_t)BKK * 4);              // 32 KiB
  bf16_t* w1b   = (bf16_t*)take((size_t)DFFN * DD * 2);       // 1 MiB
  bf16_t* w2b   = (bf16_t*)take((size_t)DD * DFFN * 2);       // 1 MiB
  float*  u     = (float*)take((size_t)BKK * DD * 4);         // 8 MiB
  float*  t     = (float*)take((size_t)BKK * DD * 4);         // 8 MiB
  bf16_t* tb    = (bf16_t*)take((size_t)BKK * DD * 2);        // 4 MiB
  bf16_t* hbuf  = (bf16_t*)take((size_t)BKK * DFFN * 2);      // 32 MiB
  bf16_t* xtb   = (bf16_t*)take((size_t)BB * HWW * DD * 2);   // 64 MiB
  bf16_t* attnb = (bf16_t*)take((size_t)BKK * HWW * 2);       // 64 MiB
  (void)ws_size; (void)in_sizes; (void)n_in; (void)out_size;

  hipMemsetAsync(S, 0, (size_t)BKK * 4, stream);

  slotgrp_cvt_w<<<dim3((DFFN * DD + 255) / 256), dim3(256), 0, stream>>>(w1, w2, w1b, w2b);
  slotgrp_prep_tgt<<<dim3(BKK / 8), dim3(256), 0, stream>>>(tgt, tnb);
  slotgrp_xinv<<<dim3(BB, HWW / 256), dim3(256), 0, stream>>>(x, xinv);
  slotgrp_xtb<<<dim3(BB, HWW / 32, DD / 32), dim3(256), 0, stream>>>(x, xinv, xtb);
  slotgrp_dots_softmax<<<dim3(BB, HWW / 32), dim3(256), 0, stream>>>(tnb, xtb, attnb, S);
  slotgrp_tgt2<<<dim3(BB, KK / 64, DD / 64), dim3(256), 0, stream>>>(attnb, x, S, tgt, u);
  slotgrp_ln1<<<dim3(BKK), dim3(256), 0, stream>>>(u, g2, be2, t, tb);
  slotgrp_ffn1<<<dim3(BKK / 64, DFFN / 64), dim3(256), 0, stream>>>(tb, w1b, b1, hbuf);
  slotgrp_ffn2_ln<<<dim3(BKK / 32), dim3(256), 0, stream>>>(hbuf, w2b, b2, t, g3, be3, out);
}
